// GPTModel_67723044323532
// MI455X (gfx1250) — matrix operand order repaired and checked by reading, not compiled, not run on hardware
//
#include <hip/hip_runtime.h>
#include <hip/hip_bf16.h>
#include <math.h>

// ---------------------------------------------------------------------------
// GPT-2 small forward (L=6, B=2, T=1024, E=768, H=12, D=64, V=50257), gfx1250.
//  * every matmul on v_wmma_f32_16x16x32_bf16 (bf16 in, fp32 accumulate)
//  * GEMM: 128x128x32 tile, 8 waves x (32x64), double-buffered LDS staged by
//    GLOBAL_LOAD_ASYNC_TO_LDS_B128 (inline asm, ASYNCcnt) -> load/compute
//    overlap without the VGPR round-trip
//  * weights pre-transposed to (N,K) bf16 once per launch so all B-fragments
//    are contiguous ds_load_b128; QKV fused into a single N=2304 GEMM
//  * flash attention (online softmax), K staged async, V staged transposed
// ---------------------------------------------------------------------------

typedef __attribute__((ext_vector_type(16))) __bf16 v16bf;
typedef __attribute__((ext_vector_type(8)))  __bf16 v8bf;
typedef __attribute__((ext_vector_type(8)))  float  v8f;

#define E_DIM 768
#define T_SEQ 1024
#define NB    2
#define NH    12
#define DH    64
#define NL    6
#define VOCAB 50257
#define FF    3072
#define MROWS (NB * T_SEQ)   // 2048
#define LDQ   (3 * E_DIM)    // fused qkv row stride (2304)

// async copy: LDS[lds_addr .. +15] = MEM[gaddr .. +15]   (per active lane)
__device__ __forceinline__ void async_b128(uint32_t lds_addr,
                                           const void* gaddr) {
  asm volatile("global_load_async_to_lds_b128 %0, %1, off"
               :: "v"(lds_addr),
                  "v"((unsigned long long)(uintptr_t)gaddr)
               : "memory");
}
__device__ __forceinline__ void wait_async0() {
  asm volatile("s_wait_asynccnt 0" ::: "memory");
}
__device__ __forceinline__ uint32_t lds_off(const void* p) {
  return (uint32_t)(uintptr_t)p;   // generic LDS addr: low 32 bits = LDS byte
}

// -------------------------------- helpers ----------------------------------

__global__ void k_f2bf(const float* __restrict__ in, __bf16* __restrict__ out,
                       long long n) {
  long long i = (long long)blockIdx.x * blockDim.x + threadIdx.x;
  if (i < n) out[i] = (__bf16)in[i];
}

// transpose fp32 (K x N) -> bf16 (N x K), 32x32 tiles via LDS
__global__ __launch_bounds__(256) void k_tbf(const float* __restrict__ in,
                                             __bf16* __restrict__ out,
                                             int K, int N) {
  __shared__ float tile[32][33];
  int kb = blockIdx.y * 32, nb = blockIdx.x * 32;
  int tx = threadIdx.x & 31, ty = threadIdx.x >> 5;   // ty 0..7
  for (int yy = ty; yy < 32; yy += 8) {
    int k = kb + yy, n = nb + tx;
    tile[yy][tx] = (k < K && n < N) ? in[(size_t)k * N + n] : 0.f;
  }
  __syncthreads();
  for (int yy = ty; yy < 32; yy += 8) {
    int n = nb + yy, k = kb + tx;
    if (n < N && k < K) out[(size_t)n * K + k] = (__bf16)tile[tx][yy];
  }
}

__global__ void k_embed(const int* __restrict__ tok,
                        const float* __restrict__ wte,
                        const float* __restrict__ wpe,
                        float* __restrict__ x) {
  int row = blockIdx.x;            // b*T + t
  int t   = row % T_SEQ;
  int v   = tok[row];
  for (int c = threadIdx.x; c < E_DIM; c += blockDim.x)
    x[(size_t)row * E_DIM + c] =
        wte[(size_t)v * E_DIM + c] + wpe[(size_t)t * E_DIM + c];
}

__global__ __launch_bounds__(256) void k_layernorm(
    const float* __restrict__ x, const float* __restrict__ sc,
    const float* __restrict__ bi, __bf16* __restrict__ out) {
  __shared__ float r1[256], r2[256];
  int row = blockIdx.x;
  const float* xr = x + (size_t)row * E_DIM;
  float s1 = 0.f, s2 = 0.f;
  for (int c = threadIdx.x; c < E_DIM; c += 256) {
    float v = xr[c];
    s1 += v; s2 += v * v;
  }
  r1[threadIdx.x] = s1; r2[threadIdx.x] = s2;
  __syncthreads();
  for (int s = 128; s > 0; s >>= 1) {
    if (threadIdx.x < s) {
      r1[threadIdx.x] += r1[threadIdx.x + s];
      r2[threadIdx.x] += r2[threadIdx.x + s];
    }
    __syncthreads();
  }
  float mu   = r1[0] * (1.0f / E_DIM);
  float var  = r2[0] * (1.0f / E_DIM) - mu * mu;
  float rstd = rsqrtf(var + 1e-5f);
  for (int c = threadIdx.x; c < E_DIM; c += 256)
    out[(size_t)row * E_DIM + c] =
        (__bf16)((xr[c] - mu) * rstd * sc[c] + bi[c]);
}

// ------------------------------ WMMA GEMM ----------------------------------
// C[M,N] = act(A[M,K] @ Bt^T + bias) + resid ; Bt stored (N,K) row-major.
// 256 thr = 8 waves; block tile 128(M) x 128(N) x 32(K); wave tile 32x64.
// Double-buffered LDS, async staging.  Requires M%128==0, K%32==0.
__global__ __launch_bounds__(256) void k_gemm(
    const __bf16* __restrict__ A, const __bf16* __restrict__ Bt,
    int M, int N, int K,
    const float* __restrict__ bias,   // nullable [N]
    const float* __restrict__ resid,  // nullable [M,N] fp32
    float* __restrict__ outf,         // nullable
    __bf16* __restrict__ outb,        // nullable
    int act)                          // 0 none, 1 exact GELU
{
  __shared__ __attribute__((aligned(16))) __bf16 sA[2][128 * 32];  // [m][k]
  __shared__ __attribute__((aligned(16))) __bf16 sB[2][128 * 32];  // [n][k]
  const int tid  = threadIdx.x;
  const int m0   = blockIdx.y * 128;
  const int n0   = blockIdx.x * 128;
  const int lane = tid & 31;
  const int half = lane >> 4;
  const int l    = lane & 15;
  const int w    = tid >> 5;
  const int wm   = (w >> 1) * 32;    // 0,32,64,96
  const int wn   = (w & 1) * 64;     // 0,64

  // issue one 128x32 K-tile (A and B) into LDS buffer pb via async DMA
  auto issue_tile = [&](int k0, int pb) {
#pragma unroll
    for (int i = 0; i < 2; i++) {
      int c  = tid + i * 256;        // 0..511
      int r  = c >> 2;               // 0..127
      int kk = (c & 3) * 8;          // 0,8,16,24
      async_b128(lds_off(&sA[pb][r * 32 + kk]),
                 A + (size_t)(m0 + r) * K + k0 + kk);
      if (n0 + r < N)                // OOB cols masked in epilogue
        async_b128(lds_off(&sB[pb][r * 32 + kk]),
                   Bt + (size_t)(n0 + r) * K + k0 + kk);
    }
  };

  v8f acc[2][4] = {};
  const int nk = K >> 5;
  int p = 0;
  issue_tile(0, 0);

  for (int kt = 0; kt < nk; kt++) {
    wait_async0();                 // this wave's DMA for tile kt complete
    __syncthreads();               // everyone's DMA complete
    if (kt + 1 < nk) issue_tile((kt + 1) << 5, p ^ 1);

    v16bf af[2], bw[4];
#pragma unroll
    for (int i = 0; i < 2; i++) {
      const __bf16* base = &sA[p][(wm + 16 * i + l) * 32];
      v8bf lo = *(const v8bf*)(base + 16 * half);
      v8bf hi = *(const v8bf*)(base + 16 * half + 8);
      af[i] = __builtin_shufflevector(lo, hi, 0, 1, 2, 3, 4, 5, 6, 7,
                                      8, 9, 10, 11, 12, 13, 14, 15);
    }
#pragma unroll
    for (int j = 0; j < 4; j++) {
      const __bf16* base = &sB[p][(wn + 16 * j + l) * 32 + 16 * half];
      v8bf lo = *(const v8bf*)(base);
      v8bf hi = *(const v8bf*)(base + 8);
      bw[j] = __builtin_shufflevector(lo, hi, 0, 1, 2, 3, 4, 5, 6, 7,
                                      8, 9, 10, 11, 12, 13, 14, 15);
    }
#pragma unroll
    for (int i = 0; i < 2; i++)
#pragma unroll
      for (int j = 0; j < 4; j++)
        acc[i][j] = __builtin_amdgcn_wmma_f32_16x16x32_bf16(
            false, af[i], false, bw[j], (short)0, acc[i][j], false, false);
    p ^= 1;
  }

#pragma unroll
  for (int i = 0; i < 2; i++)
#pragma unroll
    for (int j = 0; j < 4; j++)
#pragma unroll
      for (int r = 0; r < 8; r++) {
        int row = m0 + wm + 16 * i + r + 8 * half;
        int col = n0 + wn + 16 * j + l;
        if (col < N) {
          float v = acc[i][j][r];
          if (bias)  v += bias[col];
          if (act == 1) v = 0.5f * v * (1.0f + erff(v * 0.70710678118f));
          if (resid) v += resid[(size_t)row * N + col];
          if (outf)  outf[(size_t)row * N + col] = v;
          if (outb)  outb[(size_t)row * N + col] = (__bf16)v;
        }
      }
}

// --------------------------- flash attention -------------------------------
// qkv: (MROWS, 3E) bf16 [Q | K | V], head h at col h*64 in each third.
// O  : (MROWS, E) bf16.  grid (T/64, B*H), 128 thr = 4 waves x 16 q-rows.
__global__ __launch_bounds__(128) void k_attn(const __bf16* __restrict__ qkv,
                                              __bf16* __restrict__ O) {
  __shared__ __attribute__((aligned(16))) __bf16 sK[32 * 64];    // [key][d]
  __shared__ __attribute__((aligned(16))) __bf16 sVt[64 * 32];   // [d][key]
  __shared__ __attribute__((aligned(16))) __bf16 sP[4][16 * 32];
  const int tid  = threadIdx.x;
  const int wv   = tid >> 5;
  const int lane = tid & 31;
  const int half = lane >> 4;
  const int l    = lane & 15;
  const int b    = blockIdx.y / NH;
  const int hh   = blockIdx.y % NH;
  const size_t rowb = (size_t)b * T_SEQ * LDQ;
  const __bf16* Qb = qkv + rowb + hh * DH;
  const __bf16* Kb = Qb + E_DIM;
  const __bf16* Vb = Qb + 2 * E_DIM;
  const int q0 = blockIdx.x * 64 + wv * 16;
  const float scale = 0.125f;      // 1/sqrt(64)

  // Q as two 16x32 A-fragments over the head dim
  v16bf qa[2];
  {
    const __bf16* qrow = Qb + (size_t)(q0 + l) * LDQ;
#pragma unroll
    for (int c = 0; c < 2; c++) {
      v8bf lo = *(const v8bf*)(qrow + 32 * c + 16 * half);
      v8bf hi = *(const v8bf*)(qrow + 32 * c + 16 * half + 8);
      qa[c] = __builtin_shufflevector(lo, hi, 0, 1, 2, 3, 4, 5, 6, 7,
                                      8, 9, 10, 11, 12, 13, 14, 15);
    }
  }

  v8f acc[4] = {};
  float mrow[8], lrow[8];
#pragma unroll
  for (int r = 0; r < 8; r++) { mrow[r] = -INFINITY; lrow[r] = 0.f; }

  const int nkb = 2 * blockIdx.x + 2;
  for (int kb = 0; kb < nkb; kb++) {
    const int klo = kb * 32;
    // K tile: async DMA, kept [key][d] (contiguous B-frags for Q@K^T)
#pragma unroll
    for (int i = 0; i < 2; i++) {
      int c  = tid + i * 128;
      int kr = c >> 3, d0 = (c & 7) * 8;
      async_b128(lds_off(&sK[kr * 64 + d0]),
                 Kb + (size_t)(klo + kr) * LDQ + d0);
    }
    // V tile: staged transposed [d][key] (contiguous B-frags for P@V)
#pragma unroll
    for (int i = 0; i < 2; i++) {
      int c  = tid + i * 128;
      int kr = c >> 3, d0 = (c & 7) * 8;
      v8bf vv = *(const v8bf*)(Vb + (size_t)(klo + kr) * LDQ + d0);
#pragma unroll
      for (int j = 0; j < 8; j++) sVt[(d0 + j) * 32 + kr] = vv[j];
    }
    wait_async0();
    __syncthreads();

    // S = Q @ K^T : two 16-key tiles
    v8f sc[2];
#pragma unroll
    for (int kt = 0; kt < 2; kt++) {
      v8f z = {};
#pragma unroll
      for (int c = 0; c < 2; c++) {
        const __bf16* base = &sK[(kt * 16 + l) * 64 + 32 * c + 16 * half];
        v8bf lo = *(const v8bf*)(base);
        v8bf hi = *(const v8bf*)(base + 8);
        v16bf kf = __builtin_shufflevector(lo, hi, 0, 1, 2, 3, 4, 5, 6, 7,
                                           8, 9, 10, 11, 12, 13, 14, 15);
        z = __builtin_amdgcn_wmma_f32_16x16x32_bf16(
            false, qa[c], false, kf, (short)0, z, false, false);
      }
      sc[kt] = z;
    }

    // online softmax (rows live in 16-lane half-waves)
#pragma unroll
    for (int r = 0; r < 8; r++) {
      int qi = q0 + r + 8 * half;
      float s0 = sc[0][r] * scale, s1 = sc[1][r] * scale;
      if (klo + l      > qi) s0 = -INFINITY;
      if (klo + 16 + l > qi) s1 = -INFINITY;
      float mx = fmaxf(s0, s1);
#pragma unroll
      for (int off = 1; off < 16; off <<= 1)
        mx = fmaxf(mx, __shfl_xor(mx, off, 32));
      float mnew  = fmaxf(mrow[r], mx);
      float alpha = (mrow[r] == -INFINITY) ? 0.f : expf(mrow[r] - mnew);
      float p0 = (s0 == -INFINITY) ? 0.f : expf(s0 - mnew);
      float p1 = (s1 == -INFINITY) ? 0.f : expf(s1 - mnew);
      float rs = p0 + p1;
#pragma unroll
      for (int off = 1; off < 16; off <<= 1) rs += __shfl_xor(rs, off, 32);
      lrow[r] = lrow[r] * alpha + rs;
      mrow[r] = mnew;
#pragma unroll
      for (int dt = 0; dt < 4; dt++) acc[dt][r] *= alpha;
      sP[wv][(r + 8 * half) * 32 + l]      = (__bf16)p0;
      sP[wv][(r + 8 * half) * 32 + 16 + l] = (__bf16)p1;
    }

    // P: C-frag -> A-frag via wave-private LDS scratch
    v16bf pa;
    {
      const __bf16* prow = &sP[wv][l * 32];
      v8bf lo = *(const v8bf*)(prow + 16 * half);
      v8bf hi = *(const v8bf*)(prow + 16 * half + 8);
      pa = __builtin_shufflevector(lo, hi, 0, 1, 2, 3, 4, 5, 6, 7,
                                   8, 9, 10, 11, 12, 13, 14, 15);
    }
    // O += P @ V
#pragma unroll
    for (int dt = 0; dt < 4; dt++) {
      const __bf16* base = &sVt[(dt * 16 + l) * 32 + 16 * half];
      v8bf lo = *(const v8bf*)(base);
      v8bf hi = *(const v8bf*)(base + 8);
      v16bf vb = __builtin_shufflevector(lo, hi, 0, 1, 2, 3, 4, 5, 6, 7,
                                         8, 9, 10, 11, 12, 13, 14, 15);
      acc[dt] = __builtin_amdgcn_wmma_f32_16x16x32_bf16(
          false, pa, false, vb, (short)0, acc[dt], false, false);
    }
    __syncthreads();
  }

#pragma unroll
  for (int dt = 0; dt < 4; dt++)
#pragma unroll
    for (int r = 0; r < 8; r++) {
      int t = q0 + r + 8 * half;
      O[(size_t)(b * T_SEQ + t) * E_DIM + hh * DH + dt * 16 + l] =
          (__bf16)(acc[dt][r] / lrow[r]);
    }
}

// ------------------------------- launch ------------------------------------

extern "C" void kernel_launch(void* const* d_in, const int* in_sizes, int n_in,
                              void* d_out, int out_size, void* d_ws,
                              size_t ws_size, hipStream_t stream) {
  (void)in_sizes; (void)n_in; (void)out_size; (void)ws_size;
  const int*   tokens = (const int*)d_in[0];
  const float* wte  = (const float*)d_in[1];
  const float* wpe  = (const float*)d_in[2];
  const float* Wq   = (const float*)d_in[3];
  const float* Wk   = (const float*)d_in[4];
  const float* Wv   = (const float*)d_in[5];
  const float* Wo   = (const float*)d_in[6];
  const float* bo   = (const float*)d_in[7];
  const float* ln1s = (const float*)d_in[8];
  const float* ln1b = (const float*)d_in[9];
  const float* W1   = (const float*)d_in[10];
  const float* b1   = (const float*)d_in[11];
  const float* W2   = (const float*)d_in[12];
  const float* b2   = (const float*)d_in[13];
  const float* ln2s = (const float*)d_in[14];
  const float* ln2b = (const float*)d_in[15];
  const float* lnfs = (const float*)d_in[16];
  const float* lnfb = (const float*)d_in[17];
  float* out = (float*)d_out;

  char* ws = (char*)d_ws;
  size_t off = 0;
  auto alloc = [&](size_t bytes) -> void* {
    void* p = ws + off;
    off = (off + bytes + 255) & ~(size_t)255;
    return p;
  };
  const size_t EE = (size_t)E_DIM * E_DIM;
  __bf16* wteb   = (__bf16*)alloc((size_t)VOCAB * E_DIM * 2);  // already (N,K)
  __bf16* WQKVt  = (__bf16*)alloc((size_t)NL * 3 * EE * 2);    // (2304,768)/layer
  __bf16* Wot    = (__bf16*)alloc((size_t)NL * EE * 2);        // (768,768)
  __bf16* W1t    = (__bf16*)alloc((size_t)NL * E_DIM * FF * 2);// (3072,768)
  __bf16* W2t    = (__bf16*)alloc((size_t)NL * E_DIM * FF * 2);// (768,3072)
  float*  x      = (float*) alloc((size_t)MROWS * E_DIM * 4);
  __bf16* hb     = (__bf16*)alloc((size_t)MROWS * E_DIM * 2);
  __bf16* qkvb   = (__bf16*)alloc((size_t)MROWS * LDQ * 2);
  __bf16* ob     = (__bf16*)alloc((size_t)MROWS * E_DIM * 2);
  __bf16* midb   = (__bf16*)alloc((size_t)MROWS * FF * 2);

  k_f2bf<<<dim3((unsigned)(((long long)VOCAB * E_DIM + 255) / 256)), 256, 0,
           stream>>>(wte, wteb, (long long)VOCAB * E_DIM);

  const dim3 t256(256);
  const dim3 gT_EE(E_DIM / 32, E_DIM / 32);     // 768x768 transpose
  const dim3 gT_1(FF / 32, E_DIM / 32);         // (768,3072) -> (3072,768)
  const dim3 gT_2(E_DIM / 32, FF / 32);         // (3072,768) -> (768,3072)
  for (int layer = 0; layer < NL; layer++) {
    __bf16* qk = WQKVt + (size_t)layer * 3 * EE;
    k_tbf<<<gT_EE, t256, 0, stream>>>(Wq + (size_t)layer * EE, qk,          E_DIM, E_DIM);
    k_tbf<<<gT_EE, t256, 0, stream>>>(Wk + (size_t)layer * EE, qk + EE,     E_DIM, E_DIM);
    k_tbf<<<gT_EE, t256, 0, stream>>>(Wv + (size_t)layer * EE, qk + 2 * EE, E_DIM, E_DIM);
    k_tbf<<<gT_EE, t256, 0, stream>>>(Wo + (size_t)layer * EE,
                                      Wot + (size_t)layer * EE, E_DIM, E_DIM);
    k_tbf<<<gT_1, t256, 0, stream>>>(W1 + (size_t)layer * E_DIM * FF,
                                     W1t + (size_t)layer * E_DIM * FF, E_DIM, FF);
    k_tbf<<<gT_2, t256, 0, stream>>>(W2 + (size_t)layer * E_DIM * FF,
                                     W2t + (size_t)layer * E_DIM * FF, FF, E_DIM);
  }

  k_embed<<<MROWS, 256, 0, stream>>>(tokens, wte, wpe, x);

  const dim3 gQKV(3 * E_DIM / 128, MROWS / 128);      // N=2304
  const dim3 gE(E_DIM / 128, MROWS / 128);            // N=768
  const dim3 gF(FF / 128, MROWS / 128);               // N=3072
  const dim3 gV((VOCAB + 127) / 128, MROWS / 128);    // N=50257
  const dim3 gA(T_SEQ / 64, NB * NH);

  for (int layer = 0; layer < NL; layer++) {
    const __bf16* wqkv = WQKVt + (size_t)layer * 3 * EE;
    const __bf16* wo   = Wot + (size_t)layer * EE;
    const __bf16* w1   = W1t + (size_t)layer * E_DIM * FF;
    const __bf16* w2   = W2t + (size_t)layer * E_DIM * FF;

    k_layernorm<<<MROWS, 256, 0, stream>>>(
        x, ln1s + (size_t)layer * E_DIM, ln1b + (size_t)layer * E_DIM, hb);

    // fused qkv = hb @ [Wq|Wk|Wv]
    k_gemm<<<gQKV, t256, 0, stream>>>(hb, wqkv, MROWS, 3 * E_DIM, E_DIM,
                                      nullptr, nullptr, nullptr, qkvb, 0);

    k_attn<<<gA, 128, 0, stream>>>(qkvb, ob);

    // x = x + ob @ Wo + bo
    k_gemm<<<gE, t256, 0, stream>>>(ob, wo, MROWS, E_DIM, E_DIM,
                                    bo + (size_t)layer * E_DIM, x, x,
                                    nullptr, 0);

    k_layernorm<<<MROWS, 256, 0, stream>>>(
        x, ln2s + (size_t)layer * E_DIM, ln2b + (size_t)layer * E_DIM, hb);

    // mid = gelu(hb @ W1 + b1)
    k_gemm<<<gF, t256, 0, stream>>>(hb, w1, MROWS, FF, E_DIM,
                                    b1 + (size_t)layer * FF, nullptr,
                                    nullptr, midb, 1);
    // x = x + mid @ W2 + b2
    k_gemm<<<gE, t256, 0, stream>>>(midb, w2, MROWS, E_DIM, FF,
                                    b2 + (size_t)layer * E_DIM, x, x,
                                    nullptr, 0);
  }

  k_layernorm<<<MROWS, 256, 0, stream>>>(x, lnfs, lnfb, hb);

  // logits = hb @ wte^T  (wte already (V,E) = (N,K))
  k_gemm<<<gV, t256, 0, stream>>>(hb, wteb, MROWS, VOCAB, E_DIM,
                                  nullptr, nullptr, out, nullptr, 0);
}
